// FlowProteinMPNN_25881472925908
// MI455X (gfx1250) — matrix operand order repaired and checked by reading, not compiled, not run on hardware
//
#include <hip/hip_runtime.h>
#include <math.h>

// ---------------------------------------------------------------------------
// FlowProteinMPNN fused layer for gfx1250 (MI455X), bf16 WMMA + f32 accum.
// B=4, L=2048, K=48, H=128, T=64.
// h_E (201 MB) is the HBM roofline term -> read exactly once, software-
// pipelined into LDS; all GEMMs via v_wmma_f32_16x16x32_bf16.
// ---------------------------------------------------------------------------

typedef __attribute__((ext_vector_type(16))) __bf16 v16bf;
typedef __attribute__((ext_vector_type(8)))  __bf16 v8bf;
typedef __attribute__((ext_vector_type(4)))  __bf16 v4bf;
typedef __attribute__((ext_vector_type(8)))  float  v8f;
typedef __attribute__((ext_vector_type(4)))  float  v4f;

#define BB 4
#define LL 2048
#define KK 48
#define HH 128
#define TT 64

__constant__ constexpr float SCALE_INV = 1.0f / 30.0f;
__constant__ constexpr float LN_EPS    = 1e-5f;

__device__ __forceinline__ __bf16 f2bf(float f) {
  return (__bf16)f;                      // native f32->bf16 cvt on gfx1250
}

__device__ __forceinline__ float gelu_exact(float x) {
  return 0.5f * x * (1.0f + erff(x * 0.70710678118654752f));
}

// A-matrix (16xK, bf16) fragment load from row-major LDS tile.
// ISA layout: lanes 0-15 hold rows 0-15 with K = kb*32 + {0..7, 16..23},
//             lanes 16-31 hold rows 0-15 with K = kb*32 + {8..15, 24..31}.
__device__ __forceinline__ v16bf load_A(const __bf16* buf, int stride, int lane, int kb) {
  const int row = lane & 15;
  const int kg  = lane >> 4;
  const __bf16* p = buf + row * stride + kb * 32 + kg * 8;
  v8bf lo = *(const v8bf*)(p);
  v8bf hi = *(const v8bf*)(p + 16);
  v16bf r;
#pragma unroll
  for (int i = 0; i < 8; ++i) { r[i] = lo[i]; r[i + 8] = hi[i]; }
  return r;
}

// B-matrix (32x16 bf16) fragment load from pre-packed weights: lane-contiguous.
__device__ __forceinline__ v16bf load_W(const __bf16* packed, int nt, int kb, int lane, int KB) {
  return *(const v16bf*)(packed + ((size_t)((nt * KB) + kb) * 32 + lane) * 16);
}

__device__ __forceinline__ v8f bf_wmma(v16bf a, v16bf b, v8f c) {
  return __builtin_amdgcn_wmma_f32_16x16x32_bf16(false, a, false, b, (short)0, c,
                                                 false, false);
}

// ---------------------------------------------------------------------------
// Prep kernel 1: repack an f32 (Kd x Nd) row-major weight into the exact
// bf16 B-matrix lane layout expected by v_wmma_f32_16x16x32_bf16, so the
// main kernel's weight loads are two contiguous b128s per fragment.
// packed index e = ((nt*KB + kb)*32 + lane)*16 + h  ->  (k,n) with
//   k = kb*32 + (h/8)*16 + (lane/16)*8 + (h%8),  n = nt*16 + (lane%16)
// ---------------------------------------------------------------------------
__global__ void pack_w_kernel(const float* __restrict__ src, __bf16* __restrict__ dst,
                              int Kd, int Nd) {
  const int KB = Kd >> 5;
  const int total = Kd * Nd;
  for (int e = blockIdx.x * blockDim.x + threadIdx.x; e < total;
       e += gridDim.x * blockDim.x) {
    const int h  = e & 15;
    const int l  = (e >> 4) & 31;
    const int kb = (e >> 9) % KB;
    const int nt = e / (KB << 9);
    const int k  = kb * 32 + ((h >> 3) << 4) + ((l >> 4) << 3) + (h & 7);
    const int n  = (nt << 4) + (l & 15);
    dst[e] = f2bf(src[k * Nd + n]);
  }
}

// ---------------------------------------------------------------------------
// Prep kernel 2: per-batch vectors.
//   t_proj[b,d] = t_emb[b]@t_proj_w + t_proj_b
//   tw1[b,d]    = t_emb[b]@W1_t + W1_b    (W1_t = rows 2H..2H+T of W1_w)
// ---------------------------------------------------------------------------
__global__ void tvec_kernel(const float* __restrict__ t_emb,
                            const float* __restrict__ t_proj_w,
                            const float* __restrict__ t_proj_b,
                            const float* __restrict__ W1_w,
                            const float* __restrict__ W1_b,
                            float* __restrict__ t_proj, float* __restrict__ tw1) {
  const int b = blockIdx.x, d = threadIdx.x;
  float a0 = t_proj_b[d];
  float a1 = W1_b[d];
  for (int t = 0; t < TT; ++t) {
    const float e = t_emb[b * TT + t];
    a0 += e * t_proj_w[t * HH + d];
    a1 += e * W1_w[(2 * HH + t) * HH + d];
  }
  t_proj[b * HH + d] = a0;
  tw1[b * HH + d]    = a1;
}

// ---------------------------------------------------------------------------
// Main fused kernel. One workgroup = 16 l-rows of one batch.
// 8 waves; wave j owns output columns [16j, 16j+16).
// Edge pipeline uses a 3-buffer LDS rotation (3 barriers/tile) and a
// register-level prefetch of the next h_E tile to overlap HBM with WMMA.
// ---------------------------------------------------------------------------
__global__ __launch_bounds__(256)
void mpnn_main(const float* __restrict__ h_V, const float* __restrict__ h_E,
               const float* __restrict__ mask_V,
               const float* __restrict__ W2_b, const float* __restrict__ W3_b,
               const float* __restrict__ ln1_g, const float* __restrict__ ln1_b,
               const float* __restrict__ ln2_g, const float* __restrict__ ln2_b,
               const float* __restrict__ ffn_in_b, const float* __restrict__ ffn_out_b,
               const float* __restrict__ t_proj, const float* __restrict__ tw1,
               const __bf16* __restrict__ pW1v, const __bf16* __restrict__ pW1e,
               const __bf16* __restrict__ pW2,  const __bf16* __restrict__ pW3,
               const __bf16* __restrict__ pFin, const __bf16* __restrict__ pFout,
               float* __restrict__ out) {
  __shared__ __bf16 Ab0[16 * HH];        // h_E tile / FFN input (bf16 A)
  __shared__ __bf16 Ab1[16 * HH];        // act1
  __shared__ __bf16 Ab2[16 * HH];        // act2
  __shared__ float  ntb[16 * HH];        // node_term (f32)
  __shared__ float  dhb[16 * HH];        // dh, later h + dh2
  __shared__ float  hb [16 * HH];        // h after LN1 (f32, residual)
  __shared__ __bf16 a3 [16 * 4 * HH];    // FFN hidden activations (bf16)

  const int tid  = threadIdx.x;
  const int wave = tid >> 5;
  const int lane = tid & 31;
  const int b    = blockIdx.x / (LL / 16);
  const int l0   = (blockIdx.x % (LL / 16)) * 16;

  const int nloc = lane & 15;
  const int col  = wave * 16 + nloc;     // output column this lane produces
  const int drow = (lane >> 4) << 3;     // D fragment: lanes16-31 -> rows +8

  const float w2b = W2_b[col];
  const float w3b = W3_b[col];
  const float fob = ffn_out_b[col];

  // Edge-pipeline weights live in registers for all 48 M-tiles.
  v16bf w1e[4], w2[4], w3[4];
#pragma unroll
  for (int kb = 0; kb < 4; ++kb) {
    w1e[kb] = load_W(pW1e, wave, kb, lane, 4);
    w2[kb]  = load_W(pW2,  wave, kb, lane, 4);
    w3[kb]  = load_W(pW3,  wave, kb, lane, 4);
  }

  // staging geometry: thread -> (row, 8 contiguous cols)
  const int srow = tid >> 4;
  const int c0   = (tid & 15) * 8;
  const float* hE_base =
      h_E + (((size_t)(b * LL + l0)) * KK + srow) * HH + c0;   // lloc=0, kt=0

  // ---- node term: (h_V + t_proj) @ W1_v  + (t_emb@W1_t + W1_b) -------------
  {
    const float* src = h_V + ((size_t)(b * LL + l0 + srow)) * HH + c0;
    const float* tp  = t_proj + b * HH + c0;
    v8bf pk;
#pragma unroll
    for (int i = 0; i < 8; ++i) pk[i] = f2bf(src[i] + tp[i]);
    *(v8bf*)(Ab0 + srow * HH + c0) = pk;
  }
  __syncthreads();

  // prefetch edge tile 0 while node GEMM runs
  v4f pfA = *(const v4f*)(hE_base);
  v4f pfB = *(const v4f*)(hE_base + 4);

  {
    v8f c = {};
#pragma unroll
    for (int kb = 0; kb < 4; ++kb)
      c = bf_wmma(load_A(Ab0, HH, lane, kb), load_W(pW1v, wave, kb, lane, 4), c);
    const float add = tw1[b * HH + col];
#pragma unroll
    for (int r = 0; r < 8; ++r)
      ntb[(r + drow) * HH + col] = c[r] + add;
  }
  __syncthreads();

  // ---- edge message pipeline: 48 M-tiles (tile t: l_loc = t/3, k = t%3) ----
  float dh3 = 0.0f;
  for (int t = 0; t < KK; ++t) {
    const int lloc = t / 3;
    const int kt   = t % 3;

    // stage prefetched h_E tile (bf16) into Ab0
    {
      v8bf pk;
#pragma unroll
      for (int i = 0; i < 4; ++i) { pk[i] = f2bf(pfA[i]); pk[i + 4] = f2bf(pfB[i]); }
      *(v8bf*)(Ab0 + srow * HH + c0) = pk;
    }
    __syncthreads();                                           // B1

    // issue global loads for tile t+1; they complete under the 12 WMMAs below
    if (t + 1 < KK) {
      const int nl = (t + 1) / 3, nk = (t + 1) % 3;
      const float* p = hE_base + ((size_t)nl * KK + nk * 16) * HH;
      pfA = *(const v4f*)(p);
      pfB = *(const v4f*)(p + 4);
    }

    // GEMM1: pre = h_E @ W1_e + node_term ; act1 = gelu(pre) -> Ab1
    {
      v8f c = {};
#pragma unroll
      for (int kb = 0; kb < 4; ++kb)
        c = bf_wmma(load_A(Ab0, HH, lane, kb), w1e[kb], c);
      const float nt = ntb[lloc * HH + col];
#pragma unroll
      for (int r = 0; r < 8; ++r)
        Ab1[(r + drow) * HH + col] = f2bf(gelu_exact(c[r] + nt));
    }
    __syncthreads();                                           // B2

    // GEMM2: act2 = gelu(act1 @ W2 + b2) -> Ab2
    {
      v8f c = {};
#pragma unroll
      for (int kb = 0; kb < 4; ++kb)
        c = bf_wmma(load_A(Ab1, HH, lane, kb), w2[kb], c);
#pragma unroll
      for (int r = 0; r < 8; ++r)
        Ab2[(r + drow) * HH + col] = f2bf(gelu_exact(c[r] + w2b));
    }
    __syncthreads();                                           // B3

    // GEMM3: msg = act2 @ W3 + b3 ; reduce rows into dh (no trailing barrier:
    // 3-buffer rotation keeps >=2 barriers between any conflicting access)
    {
      v8f c = {};
#pragma unroll
      for (int kb = 0; kb < 4; ++kb)
        c = bf_wmma(load_A(Ab2, HH, lane, kb), w3[kb], c);
      float s = 0.0f;
#pragma unroll
      for (int r = 0; r < 8; ++r) s += c[r] + w3b;
      s += __shfl_xor(s, 16, 32);        // fold rows 8..15 half onto 0..7 half
      dh3 += s;
      if (kt == 2) {
        if (lane < 16) dhb[lloc * HH + col] = dh3;
        dh3 = 0.0f;
      }
    }
  }
  __syncthreads();

  // ---- LN1: h = LN(h_V + dh/SCALE); wave handles rows 2w, 2w+1 -------------
#pragma unroll
  for (int rr = wave * 2; rr < wave * 2 + 2; ++rr) {
    const int cc = lane * 4;
    v4f hv = *(const v4f*)(h_V + ((size_t)(b * LL + l0 + rr)) * HH + cc);
    v4f dv = *(const v4f*)(dhb + rr * HH + cc);
    float v[4];
    float sum = 0.0f;
#pragma unroll
    for (int i = 0; i < 4; ++i) { v[i] = hv[i] + dv[i] * SCALE_INV; sum += v[i]; }
#pragma unroll
    for (int off = 16; off >= 1; off >>= 1) sum += __shfl_xor(sum, off, 32);
    const float mean = sum * (1.0f / HH);
    float var = 0.0f;
#pragma unroll
    for (int i = 0; i < 4; ++i) { const float d = v[i] - mean; var += d * d; }
#pragma unroll
    for (int off = 16; off >= 1; off >>= 1) var += __shfl_xor(var, off, 32);
    const float rs = rsqrtf(var * (1.0f / HH) + LN_EPS);
    v4f g  = *(const v4f*)(ln1_g + cc);
    v4f be = *(const v4f*)(ln1_b + cc);
    v4bf pk;
#pragma unroll
    for (int i = 0; i < 4; ++i) {
      const float y = (v[i] - mean) * rs * g[i] + be[i];
      hb[rr * HH + cc + i] = y;
      pk[i] = f2bf(y);
    }
    *(v4bf*)(Ab0 + rr * HH + cc) = pk;
  }
  __syncthreads();

  // ---- FFN in: (16x128) @ (128x512), gelu -> a3 ----------------------------
#pragma unroll
  for (int q = 0; q < 4; ++q) {
    const int nt = wave + q * 8;         // N-tile 0..31
    v8f c = {};
#pragma unroll
    for (int kb = 0; kb < 4; ++kb)
      c = bf_wmma(load_A(Ab0, HH, lane, kb), load_W(pFin, nt, kb, lane, 4), c);
    const float bia = ffn_in_b[nt * 16 + nloc];
#pragma unroll
    for (int r = 0; r < 8; ++r)
      a3[(r + drow) * (4 * HH) + nt * 16 + nloc] = f2bf(gelu_exact(c[r] + bia));
  }
  __syncthreads();

  // ---- FFN out: (16x512) @ (512x128); h2 = h + dh2 -> dhb ------------------
  {
    v8f c = {};
#pragma unroll
    for (int kb = 0; kb < 16; ++kb)
      c = bf_wmma(load_A(a3, 4 * HH, lane, kb), load_W(pFout, wave, kb, lane, 16), c);
#pragma unroll
    for (int r = 0; r < 8; ++r) {
      const int row = r + drow;
      dhb[row * HH + col] = hb[row * HH + col] + c[r] + fob;
    }
  }
  __syncthreads();

  // ---- LN2 + mask + store --------------------------------------------------
#pragma unroll
  for (int rr = wave * 2; rr < wave * 2 + 2; ++rr) {
    const int cc = lane * 4;
    v4f dv = *(const v4f*)(dhb + rr * HH + cc);
    float v[4];
    float sum = 0.0f;
#pragma unroll
    for (int i = 0; i < 4; ++i) { v[i] = dv[i]; sum += v[i]; }
#pragma unroll
    for (int off = 16; off >= 1; off >>= 1) sum += __shfl_xor(sum, off, 32);
    const float mean = sum * (1.0f / HH);
    float var = 0.0f;
#pragma unroll
    for (int i = 0; i < 4; ++i) { const float d = v[i] - mean; var += d * d; }
#pragma unroll
    for (int off = 16; off >= 1; off >>= 1) var += __shfl_xor(var, off, 32);
    const float rs = rsqrtf(var * (1.0f / HH) + LN_EPS);
    v4f g  = *(const v4f*)(ln2_g + cc);
    v4f be = *(const v4f*)(ln2_b + cc);
    const float m = mask_V[b * LL + l0 + rr];
    v4f o;
#pragma unroll
    for (int i = 0; i < 4; ++i)
      o[i] = m * ((v[i] - mean) * rs * g[i] + be[i]);
    *(v4f*)(out + ((size_t)(b * LL + l0 + rr)) * HH + cc) = o;
  }
}

// ---------------------------------------------------------------------------
extern "C" void kernel_launch(void* const* d_in, const int* in_sizes, int n_in,
                              void* d_out, int out_size, void* d_ws, size_t ws_size,
                              hipStream_t stream) {
  (void)in_sizes; (void)n_in; (void)out_size; (void)ws_size;

  const float* h_V       = (const float*)d_in[0];
  const float* h_E       = (const float*)d_in[1];
  /* E_idx (d_in[2]) is unused by the reference */
  const float* t_emb     = (const float*)d_in[3];
  const float* mask_V    = (const float*)d_in[4];
  const float* t_proj_w  = (const float*)d_in[5];
  const float* t_proj_b  = (const float*)d_in[6];
  const float* W1_w      = (const float*)d_in[7];
  const float* W1_b      = (const float*)d_in[8];
  const float* W2_w      = (const float*)d_in[9];
  const float* W2_b      = (const float*)d_in[10];
  const float* W3_w      = (const float*)d_in[11];
  const float* W3_b      = (const float*)d_in[12];
  const float* ln1_g     = (const float*)d_in[13];
  const float* ln1_b     = (const float*)d_in[14];
  const float* ln2_g     = (const float*)d_in[15];
  const float* ln2_b     = (const float*)d_in[16];
  const float* ffn_in_w  = (const float*)d_in[17];
  const float* ffn_in_b  = (const float*)d_in[18];
  const float* ffn_out_w = (const float*)d_in[19];
  const float* ffn_out_b = (const float*)d_in[20];
  float* out = (float*)d_out;

  // workspace layout (bf16 packed weights + per-batch f32 vectors)
  __bf16* pW1v = (__bf16*)d_ws;                 // 128*128
  __bf16* pW1e = pW1v + 128 * 128;
  __bf16* pW2  = pW1e + 128 * 128;
  __bf16* pW3  = pW2  + 128 * 128;
  __bf16* pFin = pW3  + 128 * 128;              // 128*512
  __bf16* pFout= pFin + 128 * 512;              // 512*128
  float*  tpj  = (float*)(pFout + 512 * 128);   // B*128
  float*  tw1  = tpj + BB * HH;                 // B*128

  pack_w_kernel<<<64, 256, 0, stream>>>(W1_w,             pW1v, 128, 128);
  pack_w_kernel<<<64, 256, 0, stream>>>(W1_w + 128 * 128, pW1e, 128, 128);
  pack_w_kernel<<<64, 256, 0, stream>>>(W2_w,             pW2,  128, 128);
  pack_w_kernel<<<64, 256, 0, stream>>>(W3_w,             pW3,  128, 128);
  pack_w_kernel<<<128, 256, 0, stream>>>(ffn_in_w,        pFin, 128, 512);
  pack_w_kernel<<<128, 256, 0, stream>>>(ffn_out_w,       pFout, 512, 128);

  tvec_kernel<<<BB, HH, 0, stream>>>(t_emb, t_proj_w, t_proj_b, W1_w, W1_b, tpj, tw1);

  mpnn_main<<<BB * (LL / 16), 256, 0, stream>>>(
      h_V, h_E, mask_V, W2_b, W3_b, ln1_g, ln1_b, ln2_g, ln2_b,
      ffn_in_b, ffn_out_b, tpj, tw1, pW1v, pW1e, pW2, pW3, pFin, pFout, out);
}
